// L1KnowledgeMoE_52750788329560
// MI455X (gfx1250) — compile-verified, matrix operand order checked by reading, not run on hardware
//
#include <hip/hip_runtime.h>

#define D_MODEL   2048
#define N_EXPERTS 64
#define D_HID     512
#define N_TOK     4096
#define CAP       4096       // max tokens per expert (top-2 distinct experts => <= N_TOK)
#define LN_EPS    1e-5f
#define TILE_M    32         // tokens per expert tile (2 WMMA M-tiles)

typedef __bf16 bf16;
typedef bf16  bf16x16 __attribute__((ext_vector_type(16)));
typedef float f32x8   __attribute__((ext_vector_type(8)));

union AFrag { bf16x16 v; uint4 q[2]; };

__device__ __forceinline__ unsigned short f2bf_bits(float f) {
  union { float f; unsigned int u; } c; c.f = f;
  unsigned int r = c.u + 0x7FFFu + ((c.u >> 16) & 1u);   // round-to-nearest-even
  return (unsigned short)(r >> 16);
}
__device__ __forceinline__ bf16 f2bf(float f) {
  union { unsigned short s; bf16 b; } o; o.s = f2bf_bits(f);
  return o.b;
}

// Load one 16-lane-striped bf16 fragment (two contiguous 8-elem K runs per lane)
__device__ __forceinline__ AFrag ld_frag(const bf16* __restrict__ p, int k, int half8) {
  AFrag f;
  f.q[0] = *(const uint4*)(p + k + half8);
  f.q[1] = *(const uint4*)(p + k + 16 + half8);
  return f;
}

// ---------------------------------------------------------------- kernel 1
__global__ void moe_zero_counts(int* __restrict__ counts) {
  if (threadIdx.x < N_EXPERTS) counts[threadIdx.x] = 0;
}

// ---------------------------------------------------------------- kernel 2
__global__ __launch_bounds__(256)
void moe_convert_bf16(const float* __restrict__ in, unsigned short* __restrict__ out, int n4) {
  int i = blockIdx.x * 256 + threadIdx.x;
  if (i >= n4) return;
  float4 f = ((const float4*)in)[i];
  ushort4 o;
  o.x = f2bf_bits(f.x); o.y = f2bf_bits(f.y);
  o.z = f2bf_bits(f.z); o.w = f2bf_bits(f.w);
  ((ushort4*)out)[i] = o;
}

// ---------------------------------------------------------------- kernel 3
// One block per token, 64 threads (one expert each). Renormalized top-2
// softmax weights reduce to 1/(1+e^{l2-l1}) (denominator cancels).
__global__ __launch_bounds__(64)
void moe_router(const float* __restrict__ x, const float* __restrict__ Wr,
                int* __restrict__ counts, int* __restrict__ listTok,
                float* __restrict__ listW) {
  int n = blockIdx.x;
  int e = threadIdx.x;
  __shared__ float logits[N_EXPERTS];
  const float4* xp = (const float4*)(x + (size_t)n * D_MODEL);
  const float4* wp = (const float4*)(Wr + (size_t)e * D_MODEL);
  float acc = 0.f;
  for (int i = 0; i < D_MODEL / 4; ++i) {
    float4 xv = xp[i]; float4 wv = wp[i];
    acc += xv.x * wv.x + xv.y * wv.y + xv.z * wv.z + xv.w * wv.w;
  }
  logits[e] = acc;
  __syncthreads();
  if (e == 0) {
    float l1 = -INFINITY, l2 = -INFINITY; int i1 = 0, i2 = 0;
    for (int k = 0; k < N_EXPERTS; ++k) {
      float l = logits[k];
      if (l > l1)      { l2 = l1; i2 = i1; l1 = l; i1 = k; }
      else if (l > l2) { l2 = l; i2 = k; }
    }
    float w1 = 1.f / (1.f + __expf(l2 - l1));
    float w2 = 1.f - w1;
    int p1 = atomicAdd(&counts[i1], 1);
    listTok[i1 * CAP + p1] = (n << 1) | 0;  listW[i1 * CAP + p1] = w1;
    int p2 = atomicAdd(&counts[i2], 1);
    listTok[i2 * CAP + p2] = (n << 1) | 1;  listW[i2 * CAP + p2] = w2;
  }
}

// ---------------------------------------------------------------- kernel 4
struct __align__(16) MoEShared {
  bf16  H[TILE_M][D_HID];     // 32 KB : silu(x W1^T) * gate weight (bf16)
  int   tok[TILE_M];
  int   slot[TILE_M];
  float wgt[TILE_M];
  int   valid;
};

__global__ __launch_bounds__(256)
void moe_expert_ffn(const bf16* __restrict__ xbf,    // [N_TOK][D]
                    const bf16* __restrict__ W1bf,   // [E][H][D]
                    const bf16* __restrict__ W2bf,   // [E][D][H]
                    const int*  __restrict__ counts,
                    const int*  __restrict__ listTok,
                    const float* __restrict__ listW,
                    float* __restrict__ yp) {        // [2][N_TOK][D]
  __shared__ MoEShared sh;
  int tile = blockIdx.x;                   // tile-major => same-expert blocks adjacent
  int e    = blockIdx.y;
  int cnt  = counts[e];
  if (tile * TILE_M >= cnt) return;        // block-uniform: EXEC stays all-ones

  int tid = threadIdx.x;
  if (tid < TILE_M) {
    int idx = tile * TILE_M + tid;
    int ts = 0; float w = 0.f;
    if (idx < cnt) { ts = listTok[e * CAP + idx]; w = listW[e * CAP + idx]; }
    sh.tok[tid]  = ts >> 1;                // padded rows alias token 0, weight 0
    sh.slot[tid] = ts & 1;
    sh.wgt[tid]  = w;
  }
  if (tid == 0) {
    int v = cnt - tile * TILE_M;
    sh.valid = v > TILE_M ? TILE_M : v;
  }
  __syncthreads();

  int wv    = tid >> 5;                    // wave 0..7
  int lane  = tid & 31;
  int lm    = lane & 15;                   // A row / B column within 16x16 tile
  int half8 = (lane & 16) ? 8 : 0;         // K sub-block select per ISA layout

  // ---- Stage 1: H = silu(X @ W1^T) ; M=32, N=512 (64 cols/wave), K=2048 ----
  const bf16* arow0 = xbf + (size_t)sh.tok[lm]      * D_MODEL;   // M-tile 0
  const bf16* arow1 = xbf + (size_t)sh.tok[16 + lm] * D_MODEL;   // M-tile 1
  const bf16* bptr  = W1bf + ((size_t)e * D_HID + (wv * 64 + lm)) * (size_t)D_MODEL;

  f32x8 acc1[2][4];
  #pragma unroll
  for (int mt = 0; mt < 2; ++mt)
    #pragma unroll
    for (int nt = 0; nt < 4; ++nt)
      #pragma unroll
      for (int r = 0; r < 8; ++r) acc1[mt][nt][r] = 0.f;

  AFrag a0c = ld_frag(arow0, 0, half8);
  AFrag a1c = ld_frag(arow1, 0, half8);
  AFrag bc[4];
  #pragma unroll
  for (int nt = 0; nt < 4; ++nt)
    bc[nt] = ld_frag(bptr + (size_t)nt * 16 * D_MODEL, 0, half8);

  #pragma unroll 2
  for (int kc = 0; kc < 64; ++kc) {
    int kn = (kc < 63) ? (kc + 1) * 32 : 63 * 32;        // clamped tail reload
    AFrag a0n = ld_frag(arow0, kn, half8);
    AFrag a1n = ld_frag(arow1, kn, half8);
    AFrag bn[4];
    #pragma unroll
    for (int nt = 0; nt < 4; ++nt)
      bn[nt] = ld_frag(bptr + (size_t)nt * 16 * D_MODEL, kn, half8);
    __builtin_prefetch(bptr + kc * 32 + 512, 0, 1);      // stream ~1KB ahead
    #pragma unroll
    for (int nt = 0; nt < 4; ++nt) {
      acc1[0][nt] = __builtin_amdgcn_wmma_f32_16x16x32_bf16(
          false, a0c.v, false, bc[nt].v, (short)0, acc1[0][nt], false, false);
      acc1[1][nt] = __builtin_amdgcn_wmma_f32_16x16x32_bf16(
          false, a1c.v, false, bc[nt].v, (short)0, acc1[1][nt], false, false);
    }
    a0c = a0n; a1c = a1n;
    #pragma unroll
    for (int nt = 0; nt < 4; ++nt) bc[nt] = bn[nt];
  }

  // silu + fold in per-token gate weight, write bf16 H tile
  #pragma unroll
  for (int mt = 0; mt < 2; ++mt) {
    #pragma unroll
    for (int nt = 0; nt < 4; ++nt) {
      int hcol = wv * 64 + nt * 16 + lm;
      #pragma unroll
      for (int r = 0; r < 8; ++r) {
        int m = mt * 16 + half8 + r;
        float v = acc1[mt][nt][r];
        float s = v / (1.f + __expf(-v));
        sh.H[m][hcol] = f2bf(s * sh.wgt[m]);
      }
    }
  }
  __syncthreads();

  // ---- Stage 2: Y = H @ W2^T ; M=32, N=2048 (256 cols/wave), K=512 ----
  for (int nt = 0; nt < 16; ++nt) {
    int dcol = wv * 256 + nt * 16;
    const bf16* bptr2 = W2bf + ((size_t)e * D_MODEL + dcol + lm) * (size_t)D_HID;

    f32x8 acc0, acc2;
    #pragma unroll
    for (int r = 0; r < 8; ++r) { acc0[r] = 0.f; acc2[r] = 0.f; }

    AFrag h0c = ld_frag(&sh.H[lm][0],      0, half8);
    AFrag h1c = ld_frag(&sh.H[16 + lm][0], 0, half8);
    AFrag b2c = ld_frag(bptr2, 0, half8);

    #pragma unroll 2
    for (int kc = 0; kc < 16; ++kc) {
      int kn = (kc < 15) ? (kc + 1) * 32 : 15 * 32;
      AFrag h0n = ld_frag(&sh.H[lm][0],      kn, half8);
      AFrag h1n = ld_frag(&sh.H[16 + lm][0], kn, half8);
      AFrag b2n = ld_frag(bptr2, kn, half8);
      acc0 = __builtin_amdgcn_wmma_f32_16x16x32_bf16(
          false, h0c.v, false, b2c.v, (short)0, acc0, false, false);
      acc2 = __builtin_amdgcn_wmma_f32_16x16x32_bf16(
          false, h1c.v, false, b2c.v, (short)0, acc2, false, false);
      h0c = h0n; h1c = h1n; b2c = b2n;
    }

    int dd = dcol + lm;
    #pragma unroll
    for (int r = 0; r < 8; ++r) {
      int m0 = half8 + r;
      if (m0 < sh.valid)
        yp[((size_t)sh.slot[m0] * N_TOK + sh.tok[m0]) * D_MODEL + dd] = acc0[r];
      int m1 = 16 + half8 + r;
      if (m1 < sh.valid)
        yp[((size_t)sh.slot[m1] * N_TOK + sh.tok[m1]) * D_MODEL + dd] = acc2[r];
    }
  }
}

// ---------------------------------------------------------------- kernel 5
__global__ __launch_bounds__(256)
void moe_combine_ln(const float* __restrict__ yp, const float* __restrict__ gamma,
                    const float* __restrict__ beta, float* __restrict__ out) {
  int n   = blockIdx.x;
  int tid = threadIdx.x;
  const float* a = yp + (size_t)n * D_MODEL;
  const float* b = yp + (size_t)N_TOK * D_MODEL + (size_t)n * D_MODEL;
  float ys[8];
  float s = 0.f, s2 = 0.f;
  #pragma unroll
  for (int k = 0; k < 8; ++k) {
    int i = tid + k * 256;
    float y = a[i] + b[i];
    ys[k] = y; s += y; s2 += y * y;
  }
  __shared__ float rs[256], rq[256];
  rs[tid] = s; rq[tid] = s2;
  __syncthreads();
  for (int st = 128; st > 0; st >>= 1) {
    if (tid < st) { rs[tid] += rs[tid + st]; rq[tid] += rq[tid + st]; }
    __syncthreads();
  }
  float mu  = rs[0] * (1.f / D_MODEL);
  float var = rq[0] * (1.f / D_MODEL) - mu * mu;
  float inv = rsqrtf(var + LN_EPS);
  #pragma unroll
  for (int k = 0; k < 8; ++k) {
    int i = tid + k * 256;
    out[(size_t)n * D_MODEL + i] = (ys[k] - mu) * inv * gamma[i] + beta[i];
  }
  if (n == 0 && tid == 0) out[(size_t)N_TOK * D_MODEL] = 0.f;  // aux loss = 0
}

// ---------------------------------------------------------------- launch
extern "C" void kernel_launch(void* const* d_in, const int* in_sizes, int n_in,
                              void* d_out, int out_size, void* d_ws, size_t ws_size,
                              hipStream_t stream) {
  (void)in_sizes; (void)n_in; (void)out_size; (void)ws_size;
  const float* x     = (const float*)d_in[0];
  const float* Wr    = (const float*)d_in[1];
  const float* W1    = (const float*)d_in[2];
  const float* W2    = (const float*)d_in[3];
  const float* gamma = (const float*)d_in[4];
  const float* beta  = (const float*)d_in[5];
  float* out = (float*)d_out;

  char* ws = (char*)d_ws;
  size_t off = 0;
  auto wsalloc = [&](size_t bytes) -> void* {
    void* p = ws + off;
    off = (off + bytes + 255) & ~(size_t)255;
    return p;
  };
  int*   counts  = (int*)  wsalloc((size_t)N_EXPERTS * 4);
  int*   listTok = (int*)  wsalloc((size_t)N_EXPERTS * CAP * 4);
  float* listW   = (float*)wsalloc((size_t)N_EXPERTS * CAP * 4);
  unsigned short* xbf  = (unsigned short*)wsalloc((size_t)N_TOK * D_MODEL * 2);
  unsigned short* W1bf = (unsigned short*)wsalloc((size_t)N_EXPERTS * D_HID * D_MODEL * 2);
  unsigned short* W2bf = (unsigned short*)wsalloc((size_t)N_EXPERTS * D_MODEL * D_HID * 2);
  float* yp = (float*)wsalloc((size_t)2 * N_TOK * D_MODEL * 4);

  moe_zero_counts<<<1, 64, 0, stream>>>(counts);

  const int nx4 = (N_TOK * D_MODEL) / 4;                       // 2,097,152
  const int nw4 = (N_EXPERTS * D_HID * D_MODEL) / 4;           // 16,777,216
  moe_convert_bf16<<<nx4 / 256, 256, 0, stream>>>(x,  xbf,  nx4);
  moe_convert_bf16<<<nw4 / 256, 256, 0, stream>>>(W1, W1bf, nw4);
  moe_convert_bf16<<<nw4 / 256, 256, 0, stream>>>(W2, W2bf, nw4);

  moe_router<<<N_TOK, 64, 0, stream>>>(x, Wr, counts, listTok, listW);

  dim3 eg(CAP / TILE_M, N_EXPERTS);   // (128 tiles, 64 experts); empty tiles exit
  moe_expert_ffn<<<eg, 256, 0, stream>>>((const bf16*)xbf, (const bf16*)W1bf,
                                         (const bf16*)W2bf, counts, listTok,
                                         listW, yp);

  moe_combine_ln<<<N_TOK, 256, 0, stream>>>(yp, gamma, beta, out);
}
